// MOE_24034636989179
// MI455X (gfx1250) — compile-verified, matrix operand order checked by reading, not run on hardware
//
#include <hip/hip_runtime.h>

// ---------------------------------------------------------------------------
// MoE top-2 (SwiGLU experts) for MI455X / gfx1250.
// fp32 inputs -> bf16 WMMA (v_wmma_f32_16x16x32_bf16), f32 accumulate.
// Only selected (token, expert) pairs are computed (gathered GEMMs).
// Double-buffered LDS, single barrier per K-slice; A-tiles staged with
// GLOBAL_LOAD_ASYNC_TO_LDS_B128 (ASYNCcnt), B-tiles prefetched one slice
// ahead and stored as packed bf16 pairs (v_cvt_pk_bf16_f32 + ds_store_b32).
// ---------------------------------------------------------------------------

#define T_TOK  8192
#define DMODEL 1024
#define DFFN   4096
#define NEXP   8

#define BM 128          // block tile rows (tokens)
#define BN 64           // block tile cols
#define BK 32           // K slice
#define LDP 40          // padded LDS row stride (bf16): 32 + 8 pad, keeps
                        // ds_load_b128 16B-aligned and spreads banks

typedef __bf16 bf16;
typedef __attribute__((ext_vector_type(16))) bf16  v16bf;
typedef __attribute__((ext_vector_type(8)))  bf16  v8bf;
typedef __attribute__((ext_vector_type(2)))  bf16  v2bf;
typedef __attribute__((ext_vector_type(8)))  float v8f;
typedef __attribute__((ext_vector_type(4)))  float v4f;
typedef __attribute__((ext_vector_type(4)))  int   v4i;

#if defined(__HIP_DEVICE_COMPILE__) && \
    __has_builtin(__builtin_amdgcn_global_load_async_to_lds_b128) && \
    __has_builtin(__builtin_amdgcn_s_wait_asynccnt)
#define MOE_ASYNC 1
#else
#define MOE_ASYNC 0
#endif

#if MOE_ASYNC
typedef __attribute__((address_space(1))) v4i g_v4i;
typedef __attribute__((address_space(3))) v4i l_v4i;
static __device__ __forceinline__ void async_copy16(const bf16* g, bf16* l) {
  g_v4i* gp = (g_v4i*)(unsigned long long)(uintptr_t)g;
  l_v4i* lp = (l_v4i*)(unsigned int)(uintptr_t)l;
  __builtin_amdgcn_global_load_async_to_lds_b128(gp, lp, 0, 0);
}
static __device__ __forceinline__ void async_fence() {
  __builtin_amdgcn_s_wait_asynccnt(0);
}
#else
static __device__ __forceinline__ void async_fence() {}
#endif

static __device__ __forceinline__ v8f wmma_bf16(v16bf a, v16bf b, v8f c) {
  return __builtin_amdgcn_wmma_f32_16x16x32_bf16(false, a, false, b,
                                                 (short)0, c, false, false);
}

// Load a 16x32 bf16 fragment from LDS in the WMMA register layout:
// lane&15 selects the row, (lane>>4)*8 selects the K half; two 8-elem chunks.
static __device__ __forceinline__ v16bf load_frag(const bf16* p, int row, int kb) {
  const bf16* base = p + row * LDP;
  v8bf lo = *(const v8bf*)(base + kb);
  v8bf hi = *(const v8bf*)(base + kb + 16);
  return __builtin_shufflevector(lo, hi, 0, 1, 2, 3, 4, 5, 6, 7,
                                 8, 9, 10, 11, 12, 13, 14, 15);
}

// ---------------------------------------------------------------------------
// x (fp32) -> xb (bf16), once. 8 elements / thread.
// ---------------------------------------------------------------------------
__global__ __launch_bounds__(256)
void moe_cast_x(const float* __restrict__ x, bf16* __restrict__ xb) {
  const size_t i = ((size_t)blockIdx.x * 256 + threadIdx.x) * 8;
  v4f a = *(const v4f*)(x + i);
  v4f b = *(const v4f*)(x + i + 4);
  v8bf o;
#pragma unroll
  for (int j = 0; j < 4; ++j) { o[j] = (bf16)a[j]; o[j + 4] = (bf16)b[j]; }
  *(v8bf*)(xb + i) = o;
}

// ---------------------------------------------------------------------------
// Router: logits = x@Wr + br, top-2 softmax, combine weights + expert lists.
// One wave per token.
// ---------------------------------------------------------------------------
__global__ __launch_bounds__(256)
void moe_router(const float* __restrict__ x, const float* __restrict__ Wr,
                const float* __restrict__ br, float* __restrict__ cw,
                int* __restrict__ counts, int* __restrict__ lists) {
  const int lane = threadIdx.x & 31;
  const int wave = threadIdx.x >> 5;
  const int t = blockIdx.x * 8 + wave;

  float acc[NEXP];
#pragma unroll
  for (int e = 0; e < NEXP; ++e) acc[e] = 0.f;

  const float* xp = x + (size_t)t * DMODEL;
  for (int it = 0; it < DMODEL / 32; ++it) {
    const int d = it * 32 + lane;
    const float xv = xp[d];
    const float* wr = Wr + d * NEXP;
#pragma unroll
    for (int e = 0; e < NEXP; ++e) acc[e] += xv * wr[e];
  }
#pragma unroll
  for (int off = 16; off >= 1; off >>= 1) {
#pragma unroll
    for (int e = 0; e < NEXP; ++e) acc[e] += __shfl_down(acc[e], off, 32);
  }

  if (lane == 0) {
    float lg[NEXP];
#pragma unroll
    for (int e = 0; e < NEXP; ++e) lg[e] = acc[e] + br[e];
    int e0 = 0;
#pragma unroll
    for (int e = 1; e < NEXP; ++e) if (lg[e] > lg[e0]) e0 = e;
    int e1 = (e0 == 0) ? 1 : 0;
#pragma unroll
    for (int e = 0; e < NEXP; ++e) if (e != e0 && lg[e] > lg[e1]) e1 = e;

    const float r = __expf(lg[e1] - lg[e0]);   // <= 1
    const float w0 = 1.f / (1.f + r);
    const float w1 = r * w0;

#pragma unroll
    for (int e = 0; e < NEXP; ++e) cw[t * NEXP + e] = 0.f;
    cw[t * NEXP + e0] = w0;
    cw[t * NEXP + e1] = w1;

    int p0 = atomicAdd(&counts[e0], 1);
    lists[e0 * T_TOK + p0] = t;
    int p1 = atomicAdd(&counts[e1], 1);
    lists[e1 * T_TOK + p1] = t;
  }
}

// ---------------------------------------------------------------------------
// GEMM A (per expert): H = silu(Xg@W1 + b1) * (Xg@W2 + b2).
// M = count[e], N = DFF, K = DMODEL. Dual-B shares the A tile.
// 8 waves: 4 over M x 2 over N, 32x32 per wave (8 WMMAs / K-slice / wave).
// ---------------------------------------------------------------------------
__global__ __launch_bounds__(256)
void moe_gate_up(const bf16* __restrict__ xb,
                 const float* __restrict__ W1, const float* __restrict__ b1,
                 const float* __restrict__ W2, const float* __restrict__ b2,
                 const int* __restrict__ counts, const int* __restrict__ lists,
                 bf16* __restrict__ H, int e) {
  const int cnt = counts[e];
  const int rowBase = blockIdx.y * BM;
  if (rowBase >= cnt) return;
  const int colBase = blockIdx.x * BN;

  __shared__ __align__(16) bf16 sA[2][BM * LDP];
  __shared__ __align__(16) bf16 sB1[2][BN * LDP];
  __shared__ __align__(16) bf16 sB2[2][BN * LDP];

  const int tid  = threadIdx.x;
  const int lane = tid & 31;
  const int wave = tid >> 5;
  const int waveM = wave & 3;
  const int waveN = wave >> 2;

  // A staging: 128x32 bf16 = 512 x 16B chunks; thread covers chunks tid, tid+256
  const int a_r0 = tid >> 2;            // 0..63
  const int a_r1 = a_r0 + 64;           // 64..127
  const int a_k8 = (tid & 3) * 8;       // 0,8,16,24
  int tok0, tok1;
  { const int rg0 = rowBase + a_r0; tok0 = (rg0 < cnt) ? lists[e * T_TOK + rg0] : 0;
    const int rg1 = rowBase + a_r1; tok1 = (rg1 < cnt) ? lists[e * T_TOK + rg1] : 0; }

  // B staging: 32(K) x 64(N) fp32 -> bf16 transposed to [N][K].
  // Thread covers K pair (b_k2, b_k2+1) x 4 N cols -> packed v2bf b32 stores.
  const int b_n4 = tid & 15;            // n = b_n4*4
  const int b_k2 = (tid >> 4) * 2;      // k0 = b_k2

  const float* W1p = W1 + (size_t)e * DMODEL * DFFN;
  const float* W2p = W2 + (size_t)e * DMODEL * DFFN;

  v8f acc1[2][2], acc2[2][2];
  const v8f vzero = {};
#pragma unroll
  for (int m = 0; m < 2; ++m)
#pragma unroll
    for (int n = 0; n < 2; ++n) { acc1[m][n] = vzero; acc2[m][n] = vzero; }

  v4f rB1[2], rB2[2];                   // [j] = K row b_k2 + j
#if !MOE_ASYNC
  v8bf rA0, rA1;
#endif

  auto issueA = [&](int p, int kk) {
#if MOE_ASYNC
    async_copy16(xb + (size_t)tok0 * DMODEL + kk + a_k8, &sA[p][a_r0 * LDP + a_k8]);
    async_copy16(xb + (size_t)tok1 * DMODEL + kk + a_k8, &sA[p][a_r1 * LDP + a_k8]);
#else
    rA0 = *(const v8bf*)(xb + (size_t)tok0 * DMODEL + kk + a_k8);
    rA1 = *(const v8bf*)(xb + (size_t)tok1 * DMODEL + kk + a_k8);
#endif
  };
  auto loadB = [&](int kk) {
#pragma unroll
    for (int j = 0; j < 2; ++j) {
      const size_t g = (size_t)(kk + b_k2 + j) * DFFN + colBase + b_n4 * 4;
      rB1[j] = *(const v4f*)(W1p + g);
      rB2[j] = *(const v4f*)(W2p + g);
    }
  };
  auto storeStaged = [&](int p) {
#if !MOE_ASYNC
    *(v8bf*)&sA[p][a_r0 * LDP + a_k8] = rA0;
    *(v8bf*)&sA[p][a_r1 * LDP + a_k8] = rA1;
#endif
#pragma unroll
    for (int i = 0; i < 4; ++i) {
      v2bf t1, t2;
      t1[0] = (bf16)rB1[0][i]; t1[1] = (bf16)rB1[1][i];
      t2[0] = (bf16)rB2[0][i]; t2[1] = (bf16)rB2[1][i];
      *(v2bf*)&sB1[p][(b_n4 * 4 + i) * LDP + b_k2] = t1;
      *(v2bf*)&sB2[p][(b_n4 * 4 + i) * LDP + b_k2] = t2;
    }
  };

  // prologue: stage slice 0 into buffer 0
  issueA(0, 0);
  loadB(0);
  storeStaged(0);
  async_fence();
  __syncthreads();

  const int fr = lane & 15;
  const int kb = (lane >> 4) * 8;
  int p = 0;
  for (int kk = 0; kk < DMODEL; kk += BK) {
    const bool notLast = (kk + BK < DMODEL);
    if (notLast) { issueA(p ^ 1, kk + BK); loadB(kk + BK); }

    v16bf af[2], f1[2], f2[2];
#pragma unroll
    for (int m = 0; m < 2; ++m)
      af[m] = load_frag(&sA[p][0], waveM * 32 + m * 16 + fr, kb);
#pragma unroll
    for (int n = 0; n < 2; ++n) {
      f1[n] = load_frag(&sB1[p][0], waveN * 32 + n * 16 + fr, kb);
      f2[n] = load_frag(&sB2[p][0], waveN * 32 + n * 16 + fr, kb);
    }
#pragma unroll
    for (int m = 0; m < 2; ++m)
#pragma unroll
      for (int n = 0; n < 2; ++n) {
        acc1[m][n] = wmma_bf16(af[m], f1[n], acc1[m][n]);
        acc2[m][n] = wmma_bf16(af[m], f2[n], acc2[m][n]);
      }

    if (notLast) {
      storeStaged(p ^ 1);
      async_fence();
      __syncthreads();
      p ^= 1;
    }
  }

  // epilogue: bias + SiLU * gate -> bf16 H
  const int rh = (lane >> 4) * 8;
  float bias1[2], bias2[2];
#pragma unroll
  for (int n = 0; n < 2; ++n) {
    const int col = colBase + waveN * 32 + n * 16 + fr;
    bias1[n] = b1[e * DFFN + col];
    bias2[n] = b2[e * DFFN + col];
  }
#pragma unroll
  for (int m = 0; m < 2; ++m) {
#pragma unroll
    for (int v = 0; v < 8; ++v) {
      const int rowg = rowBase + waveM * 32 + m * 16 + rh + v;
      if (rowg < cnt) {
#pragma unroll
        for (int n = 0; n < 2; ++n) {
          const int col = colBase + waveN * 32 + n * 16 + fr;
          const float a = acc1[m][n][v] + bias1[n];
          const float s = __builtin_amdgcn_rcpf(1.f + __expf(-a));
          const float h = (a * s) * (acc2[m][n][v] + bias2[n]);
          H[(size_t)rowg * DFFN + col] = (bf16)h;
        }
      }
    }
  }
}

// ---------------------------------------------------------------------------
// GEMM B (per expert): y[token] += (H @ W3 + b3) * cw[token, e].
// M = count[e], N = DMODEL, K = DFF. Experts serialized on the stream, so
// the read-modify-write on y is race-free.
// ---------------------------------------------------------------------------
__global__ __launch_bounds__(256)
void moe_down(const bf16* __restrict__ H, const float* __restrict__ W3,
              const float* __restrict__ b3, const float* __restrict__ cw,
              const int* __restrict__ counts, const int* __restrict__ lists,
              float* __restrict__ y, int e) {
  const int cnt = counts[e];
  const int rowBase = blockIdx.y * BM;
  if (rowBase >= cnt) return;
  const int colBase = blockIdx.x * BN;

  __shared__ __align__(16) bf16 sA[2][BM * LDP];
  __shared__ __align__(16) bf16 sB[2][BN * LDP];

  const int tid  = threadIdx.x;
  const int lane = tid & 31;
  const int wave = tid >> 5;
  const int waveM = wave & 3;
  const int waveN = wave >> 2;

  const int a_r0 = tid >> 2;
  const int a_r1 = a_r0 + 64;
  const int a_k8 = (tid & 3) * 8;
  const int b_n4 = tid & 15;
  const int b_k2 = (tid >> 4) * 2;

  const float* W3p = W3 + (size_t)e * DFFN * DMODEL;

  v8f acc[2][2];
  const v8f vzero = {};
#pragma unroll
  for (int m = 0; m < 2; ++m)
#pragma unroll
    for (int n = 0; n < 2; ++n) acc[m][n] = vzero;

  v4f rB3[2];
#if !MOE_ASYNC
  v8bf rA0, rA1;
#endif

  auto issueA = [&](int p, int kk) {
#if MOE_ASYNC
    async_copy16(H + (size_t)(rowBase + a_r0) * DFFN + kk + a_k8, &sA[p][a_r0 * LDP + a_k8]);
    async_copy16(H + (size_t)(rowBase + a_r1) * DFFN + kk + a_k8, &sA[p][a_r1 * LDP + a_k8]);
#else
    rA0 = *(const v8bf*)(H + (size_t)(rowBase + a_r0) * DFFN + kk + a_k8);
    rA1 = *(const v8bf*)(H + (size_t)(rowBase + a_r1) * DFFN + kk + a_k8);
#endif
  };
  auto loadB = [&](int kk) {
#pragma unroll
    for (int j = 0; j < 2; ++j) {
      const size_t g = (size_t)(kk + b_k2 + j) * DMODEL + colBase + b_n4 * 4;
      rB3[j] = *(const v4f*)(W3p + g);
    }
  };
  auto storeStaged = [&](int p) {
#if !MOE_ASYNC
    *(v8bf*)&sA[p][a_r0 * LDP + a_k8] = rA0;
    *(v8bf*)&sA[p][a_r1 * LDP + a_k8] = rA1;
#endif
#pragma unroll
    for (int i = 0; i < 4; ++i) {
      v2bf t;
      t[0] = (bf16)rB3[0][i]; t[1] = (bf16)rB3[1][i];
      *(v2bf*)&sB[p][(b_n4 * 4 + i) * LDP + b_k2] = t;
    }
  };

  issueA(0, 0);
  loadB(0);
  storeStaged(0);
  async_fence();
  __syncthreads();

  const int fr = lane & 15;
  const int kb = (lane >> 4) * 8;
  int p = 0;
  for (int kk = 0; kk < DFFN; kk += BK) {
    const bool notLast = (kk + BK < DFFN);
    if (notLast) { issueA(p ^ 1, kk + BK); loadB(kk + BK); }

    v16bf af[2], bfr[2];
#pragma unroll
    for (int m = 0; m < 2; ++m)
      af[m] = load_frag(&sA[p][0], waveM * 32 + m * 16 + fr, kb);
#pragma unroll
    for (int n = 0; n < 2; ++n)
      bfr[n] = load_frag(&sB[p][0], waveN * 32 + n * 16 + fr, kb);
#pragma unroll
    for (int m = 0; m < 2; ++m)
#pragma unroll
      for (int n = 0; n < 2; ++n)
        acc[m][n] = wmma_bf16(af[m], bfr[n], acc[m][n]);

    if (notLast) {
      storeStaged(p ^ 1);
      async_fence();
      __syncthreads();
      p ^= 1;
    }
  }

  // epilogue: (acc + b3) * cw -> y[token] (+=)
  const int rh = (lane >> 4) * 8;
  float b3v[2];
#pragma unroll
  for (int n = 0; n < 2; ++n)
    b3v[n] = b3[e * DMODEL + colBase + waveN * 32 + n * 16 + fr];

#pragma unroll
  for (int m = 0; m < 2; ++m) {
#pragma unroll
    for (int v = 0; v < 8; ++v) {
      const int rowg = rowBase + waveM * 32 + m * 16 + rh + v;
      if (rowg < cnt) {
        const int t = lists[e * T_TOK + rowg];
        const float c = cw[t * NEXP + e];
#pragma unroll
        for (int n = 0; n < 2; ++n) {
          const int col = colBase + waveN * 32 + n * 16 + fr;
          float* yp = y + (size_t)t * DMODEL + col;
          *yp += (acc[m][n][v] + b3v[n]) * c;
        }
      }
    }
  }
}

// ---------------------------------------------------------------------------
// Launch. Workspace layout (needs ~80.5 MB):
//   [0       , 32)       int   counts[8]
//   [1024    , +256KB)   int   lists[8][8192]
//   [...     , +256KB)   f32   cw[8192][8]
//   [...     , +64MB)    bf16  H[8192][4096]
//   [...     , +16MB)    bf16  xb[8192][1024]
// ---------------------------------------------------------------------------
extern "C" void kernel_launch(void* const* d_in, const int* in_sizes, int n_in,
                              void* d_out, int out_size, void* d_ws, size_t ws_size,
                              hipStream_t stream) {
  (void)in_sizes; (void)n_in; (void)out_size; (void)ws_size;
  const float* x  = (const float*)d_in[0];
  const float* Wr = (const float*)d_in[1];
  const float* br = (const float*)d_in[2];
  const float* W1 = (const float*)d_in[3];
  const float* b1 = (const float*)d_in[4];
  const float* W2 = (const float*)d_in[5];
  const float* b2 = (const float*)d_in[6];
  const float* W3 = (const float*)d_in[7];
  const float* b3 = (const float*)d_in[8];
  float* y = (float*)d_out;

  char* ws = (char*)d_ws;
  int*   counts = (int*)ws;
  int*   lists  = (int*)(ws + 1024);
  float* cw     = (float*)(ws + 1024 + (size_t)NEXP * T_TOK * 4);
  bf16*  H      = (bf16*)(ws + 1024 + (size_t)NEXP * T_TOK * 4
                              + (size_t)T_TOK * NEXP * 4);
  bf16*  xb     = (bf16*)((char*)H + (size_t)T_TOK * DFFN * 2);

  (void)hipMemsetAsync(counts, 0, NEXP * sizeof(int), stream);
  (void)hipMemsetAsync(y, 0, (size_t)T_TOK * DMODEL * sizeof(float), stream);

  moe_cast_x<<<(T_TOK * DMODEL) / (256 * 8), 256, 0, stream>>>(x, xb);
  moe_router<<<T_TOK / 8, 256, 0, stream>>>(x, Wr, br, cw, counts, lists);

  for (int e = 0; e < NEXP; ++e) {
    moe_gate_up<<<dim3(DFFN / BN, T_TOK / BM), 256, 0, stream>>>(
        xb, W1, b1, W2, b2, counts, lists, H, e);
    moe_down<<<dim3(DMODEL / BN, T_TOK / BM), 256, 0, stream>>>(
        H, W3, b3, cw, counts, lists, y, e);
  }
}
